// CrossJointAttention_10574209483189
// MI455X (gfx1250) — matrix operand order repaired and checked by reading, not compiled, not run on hardware
//
#include <hip/hip_runtime.h>
#include <math.h>

typedef _Float16 v16h __attribute__((ext_vector_type(16)));
typedef _Float16 v8h  __attribute__((ext_vector_type(8)));
typedef _Float16 v4h  __attribute__((ext_vector_type(4)));
typedef float    v8f  __attribute__((ext_vector_type(8)));
typedef float    v4f  __attribute__((ext_vector_type(4)));

namespace {
constexpr int kD   = 128;           // d_model
constexpr int kN   = 12;            // joints
constexpr int kH   = 4;             // heads
constexpr int kHD  = 32;            // head_dim
constexpr int kG   = 8;             // batches per workgroup
constexpr int kRows = kG * kN;      // 96 rows per tile
constexpr int kMT  = kRows / 16;    // 6 M-tiles
constexpr int kKS  = kD / 32;       // 4 K-steps of 32
constexpr int kLDH = kD + 8;        // padded LDS row stride (halves): breaks 256B bank period
}

// One 16-wide N-tile GEMM: D(96x16) = A(96x128,f16 in LDS) * W^T(128x16) + bias.
// B-fragment comes straight from global W (L2-resident), converted f32->f16 once
// per wave and reused across all M-tiles.
template <bool TO_LDS>
__device__ __forceinline__ void gemm_tile(
    const _Float16* __restrict__ sA_, const float* __restrict__ W,
    const float* __restrict__ bvec, _Float16* __restrict__ sOut,
    float* __restrict__ gOut, const int lane, const int nt) {
  const int nlo    = lane & 15;
  const int hiHalf = (lane >> 4) & 1;          // lanes 16..31 take the high K half
  const int ncol   = nt * 16 + nlo;

  // B fragments: 32x16 f16, lane holds column ncol, K-run contiguous in W[ncol][*]
  v16h bfr[kKS];
#pragma unroll
  for (int ks = 0; ks < kKS; ++ks) {
    const float* wp = W + (size_t)ncol * kD + ks * 32 + hiHalf * 16;
    const v4f w0 = *(const v4f*)(wp + 0);
    const v4f w1 = *(const v4f*)(wp + 4);
    const v4f w2 = *(const v4f*)(wp + 8);
    const v4f w3 = *(const v4f*)(wp + 12);
    v16h t;
#pragma unroll
    for (int e = 0; e < 4; ++e) {
      t[e]      = (_Float16)w0[e];
      t[e + 4]  = (_Float16)w1[e];
      t[e + 8]  = (_Float16)w2[e];
      t[e + 12] = (_Float16)w3[e];
    }
    bfr[ks] = t;
  }
  const float bias = bvec[ncol];

#pragma unroll
  for (int mt = 0; mt < kMT; ++mt) {
    const int arow = mt * 16 + nlo;            // A layout: row = lane % 16
    v8f acc = {0.f, 0.f, 0.f, 0.f, 0.f, 0.f, 0.f, 0.f};
#pragma unroll
    for (int ks = 0; ks < kKS; ++ks) {
      // A 16x32, contiguous k order used here: lanes<16 K{0..15}, lanes>=16 K{16..31} (matches B)
      const _Float16* ap = sA_ + arow * kLDH + ks * 32 + hiHalf * 16;
      const v8h lo = *(const v8h*)ap;          // ds_load_b128
      const v8h hi = *(const v8h*)(ap + 8);    // ds_load_b128
      v16h a;
#pragma unroll
      for (int e = 0; e < 8; ++e) { a[e] = lo[e]; a[e + 8] = hi[e]; }
      acc = __builtin_amdgcn_wmma_f32_16x16x32_f16(false, a, false, bfr[ks],
                                                   (short)0, acc, false, false);
    }
    // C/D layout: VGPR i -> M = i + 8*hiHalf, N = lane&15
#pragma unroll
    for (int i = 0; i < 8; ++i) {
      const int m = mt * 16 + i + (hiHalf ? 8 : 0);
      const float val = acc[i] + bias;
      if (TO_LDS) sOut[m * kLDH + ncol] = (_Float16)val;
      else        gOut[(size_t)m * kD + ncol] = val;
    }
  }
}

__global__ __launch_bounds__(256, 1) void cja_kernel(
    const float* __restrict__ x, const float* __restrict__ vis,
    const float* __restrict__ Wq, const float* __restrict__ bq,
    const float* __restrict__ Wk, const float* __restrict__ bk,
    const float* __restrict__ Wv, const float* __restrict__ bv,
    const float* __restrict__ Wo, const float* __restrict__ bo,
    const float* __restrict__ bias_scale, const float* __restrict__ adjacency,
    float* __restrict__ out) {
  __shared__ __align__(16) _Float16 sX[kRows * kLDH];  // x tile, later attn output
  __shared__ __align__(16) _Float16 sQ[kRows * kLDH];
  __shared__ __align__(16) _Float16 sK[kRows * kLDH];
  __shared__ __align__(16) _Float16 sV[kRows * kLDH];
  __shared__ float sAdj[kN * kN];
  __shared__ float sVis[kG * kN];

  const int tid  = threadIdx.x;
  const int lane = tid & 31;
  const int wave = tid >> 5;                   // 8 waves -> 8 N-tiles
  const int wg   = blockIdx.x;
  const float* xt   = x + (size_t)wg * kRows * kD;
  float*       outt = out + (size_t)wg * kRows * kD;

  // Warm weights into L2 (gfx1250 global_prefetch_b8), 256B granularity.
  for (int i = tid; i < 1024; i += 256) {
    const float* base = (i < 256) ? Wq : (i < 512) ? Wk : (i < 768) ? Wv : Wo;
    __builtin_prefetch(base + (size_t)(i & 255) * 64, 0, 1);
  }

  // ---- Phase 0: stage x tile (f32 -> f16), adjacency, visibility ----
  if (tid < kN * kN)  sAdj[tid] = adjacency[tid];
  if (tid < kG * kN)  sVis[tid] = vis[(size_t)wg * kG * kN + tid];
  for (int i = tid; i < kRows * kD / 4; i += 256) {
    const v4f f = *((const v4f*)xt + i);
    const int r = (i * 4) / kD;
    const int c = (i * 4) % kD;
    v4h hv;
    hv[0] = (_Float16)f[0]; hv[1] = (_Float16)f[1];
    hv[2] = (_Float16)f[2]; hv[3] = (_Float16)f[3];
    *(v4h*)&sX[r * kLDH + c] = hv;
  }
  __syncthreads();

  // ---- Phase 1: Q/K/V projections on WMMA ----
  gemm_tile<true>(sX, Wq, bq, sQ, nullptr, lane, wave);
  gemm_tile<true>(sX, Wk, bk, sK, nullptr, lane, wave);
  gemm_tile<true>(sX, Wv, bv, sV, nullptr, lane, wave);
  __syncthreads();

  // ---- Phase 2: tiny per-(batch,head,query) attention core on VALU ----
  const float bsc = bias_scale[0];
  constexpr float kInvSqrtHD = 0.17677669529663687f;  // 1/sqrt(32)
  for (int u = tid; u < kG * kH * kN; u += 256) {
    const int b  = u / (kH * kN);
    const int h  = (u / kN) % kH;
    const int qj = u % kN;
    const int rq = b * kN + qj;

    float qv[kHD];
    {
      const _Float16* qrow = &sQ[rq * kLDH + h * kHD];
#pragma unroll
      for (int d = 0; d < kHD; ++d) qv[d] = (float)qrow[d];
    }
    float sc[kN];
    float mx = -3.0e38f;
    for (int m = 0; m < kN; ++m) {
      const _Float16* krow = &sK[(b * kN + m) * kLDH + h * kHD];
      float dot = 0.f;
#pragma unroll
      for (int d = 0; d < kHD; ++d) dot += qv[d] * (float)krow[d];
      const float s = dot * kInvSqrtHD + sAdj[qj * kN + m] * bsc
                      - 10.f * (1.f - sVis[b * kN + m]);
      sc[m] = s;
      mx = fmaxf(mx, s);
    }
    float ssum = 0.f;
    for (int m = 0; m < kN; ++m) { sc[m] = __expf(sc[m] - mx); ssum += sc[m]; }
    const float inv = 1.f / ssum;

    float o[kHD];
#pragma unroll
    for (int d = 0; d < kHD; ++d) o[d] = 0.f;
    for (int m = 0; m < kN; ++m) {
      const float a = sc[m] * inv;
      const _Float16* vrow = &sV[(b * kN + m) * kLDH + h * kHD];
#pragma unroll
      for (int d = 0; d < kHD; ++d) o[d] += a * (float)vrow[d];
    }
    _Float16* orow = &sX[rq * kLDH + h * kHD];   // reuse x tile as attn output
#pragma unroll
    for (int d = 0; d < kHD; ++d) orow[d] = (_Float16)o[d];
  }
  __syncthreads();

  // ---- Phase 3: output projection on WMMA, f32 straight to HBM ----
  gemm_tile<false>(sX, Wo, bo, nullptr, outt, lane, wave);
}

extern "C" void kernel_launch(void* const* d_in, const int* in_sizes, int n_in,
                              void* d_out, int out_size, void* d_ws, size_t ws_size,
                              hipStream_t stream) {
  (void)n_in; (void)out_size; (void)d_ws; (void)ws_size;
  const float* x   = (const float*)d_in[0];
  const float* vis = (const float*)d_in[1];
  const float* Wq  = (const float*)d_in[2];
  const float* bq  = (const float*)d_in[3];
  const float* Wk  = (const float*)d_in[4];
  const float* bk  = (const float*)d_in[5];
  const float* Wv  = (const float*)d_in[6];
  const float* bv  = (const float*)d_in[7];
  const float* Wo  = (const float*)d_in[8];
  const float* bo  = (const float*)d_in[9];
  const float* bsc = (const float*)d_in[10];
  const float* adj = (const float*)d_in[11];
  float* out = (float*)d_out;

  const int batches = in_sizes[0] / (kN * kD);   // 16384
  const int grid = batches / kG;                 // 2048 workgroups
  cja_kernel<<<grid, 256, 0, stream>>>(x, vis, Wq, bq, Wk, bk, Wv, bv,
                                       Wo, bo, bsc, adj, out);
}
